// GraphConv_53206054863565
// MI455X (gfx1250) — matrix operand order repaired and checked by reading, not compiled, not run on hardware
//
#include <hip/hip_runtime.h>
#include <hip/hip_bf16.h>
#include <math.h>

typedef __attribute__((ext_vector_type(16))) __bf16 v16bf;
typedef __attribute__((ext_vector_type(4)))  __bf16 v4bf;
typedef __attribute__((ext_vector_type(8)))  float  v8f;

__device__ __forceinline__ v8f wmma_bf16(v16bf a, v16bf b, v8f c) {
  // 8 args: (neg_a, A, neg_b, B, c_mod, C, reuse_a, reuse_b)
  return __builtin_amdgcn_wmma_f32_16x16x32_bf16(false, a, false, b, (short)0, c, false, false);
}

__device__ __forceinline__ float gelu_exact(float x) {
  return 0.5f * x * (1.0f + erff(x * 0.70710678118654752f));
}

__device__ __forceinline__ void atomic_max_f32(float* addr, float val) {
  // float-as-int ordering trick; mmax is initialized to -inf
  if (val >= 0.0f) atomicMax((int*)addr, __float_as_int(val));
  else             atomicMin((unsigned int*)addr, __float_as_uint(val));
}

// ---- WMMA fragment loaders (CDNA5 16x16x32 bf16 lane layouts) ----
// A 16x32: M = lane&15 (both halves); K for slot i: (i<4 ? 2i : 8+2i) + 8*(lane>>4)
__device__ __forceinline__ v16bf load_a_f32row(const float* __restrict__ row, int kb, int hi) {
  v16bf a;
#pragma unroll
  for (int i = 0; i < 8; ++i) {
    int k = kb + ((i < 4) ? (2 * i) : (8 + 2 * i)) + 8 * hi;
    a[2 * i]     = (__bf16)row[k];
    a[2 * i + 1] = (__bf16)row[k + 1];
  }
  return a;
}
__device__ __forceinline__ v16bf load_a_bf16row(const __bf16* __restrict__ row, int kb, int hi) {
  v16bf a;
#pragma unroll
  for (int i = 0; i < 8; ++i) {
    int k = kb + ((i < 4) ? (2 * i) : (8 + 2 * i)) + 8 * hi;
    a[2 * i]     = row[k];
    a[2 * i + 1] = row[k + 1];
  }
  return a;
}
// B 32x16 from W^T stored [col][K]: N = lane&15; K for slot j: (j<4 ? 2j : 8+2j) + 8*(lane>>4)
__device__ __forceinline__ v16bf load_b_wt(const __bf16* __restrict__ wcol, int kb, int hi) {
  v16bf b;
#pragma unroll
  for (int j = 0; j < 8; ++j) {
    int k = kb + ((j < 4) ? (2 * j) : (8 + 2 * j)) + 8 * hi;
    b[2 * j]     = wcol[k];
    b[2 * j + 1] = wcol[k + 1];
  }
  return b;
}

// ---- small utility kernels ----
__global__ void fill_f32(float* __restrict__ p, float v, size_t n) {
  size_t i = (size_t)blockIdx.x * blockDim.x + threadIdx.x;
  if (i < n) p[i] = v;
}

__global__ void prep_wT(const float* __restrict__ w, __bf16* __restrict__ wT, int K, int C) {
  int i = blockIdx.x * blockDim.x + threadIdx.x;
  if (i >= K * C) return;
  int k = i / C, c = i % C;
  wT[(size_t)c * K + k] = (__bf16)w[(size_t)k * C + c];
}

// 4 floats -> 4 bf16 per thread (b128 in, b64 out)
__global__ void cvt_bf16_v4(const float* __restrict__ x, __bf16* __restrict__ y, size_t n4) {
  size_t i = (size_t)blockIdx.x * blockDim.x + threadIdx.x;
  if (i >= n4) return;
  float4 v = ((const float4*)x)[i];
  v4bf o = { (__bf16)v.x, (__bf16)v.y, (__bf16)v.z, (__bf16)v.w };
  ((v4bf*)y)[i] = o;
}

// ---- fused upscale GEMM: [N,768]@[768,128] x2 -> concat -> +bias -> GELU -> scatter ----
// 8 waves/block; wave = (mtile within block, 64-col block). cb 0,1 -> image, cb 2,3 -> text.
__global__ __launch_bounds__(256) void upscale_gemm(
    const float* __restrict__ img, const float* __restrict__ txt,
    const __bf16* __restrict__ wTimg, const __bf16* __restrict__ wTtxt,
    const float* __restrict__ bimg, const float* __restrict__ btxt,
    const int* __restrict__ ci, float* __restrict__ node, int N) {
  const int wave = threadIdx.x >> 5;
  const int lane = threadIdx.x & 31;
  const int r = lane & 15, hi = lane >> 4;
  const int mtile = blockIdx.x * 2 + (wave >> 2);
  const int cb = wave & 3;
  const int row = mtile * 16 + r;
  const int rowc = (row < N) ? row : 0;

  const float*  src  = (cb < 2) ? img  : txt;
  const __bf16* wT   = (cb < 2) ? wTimg : wTtxt;
  const float*  bias = (cb < 2) ? bimg : btxt;
  const int colbase = cb * 64;          // output column base (0..255)
  const int wcbase  = (cb & 1) * 64;    // column base inside the 128-wide weight
  const float* arow = src + (size_t)rowc * 768;
  const __bf16* wcol0 = wT + (size_t)(wcbase + r) * 768;   // t advances by 16 rows of W^T

  v8f acc[4] = {};
  v16bf a = load_a_f32row(arow, 0, hi);
  for (int kb = 0; kb < 768; kb += 32) {
    // group all loads for this iteration, then back-to-back WMMAs
    v16bf b0 = load_b_wt(wcol0 + (size_t)0 * 16 * 768, kb, hi);
    v16bf b1 = load_b_wt(wcol0 + (size_t)1 * 16 * 768, kb, hi);
    v16bf b2 = load_b_wt(wcol0 + (size_t)2 * 16 * 768, kb, hi);
    v16bf b3 = load_b_wt(wcol0 + (size_t)3 * 16 * 768, kb, hi);
    v16bf an = a;
    if (kb + 32 < 768) {
      __builtin_prefetch(arow + kb + 96, 0, 0);   // stream next A chunk from HBM
      an = load_a_f32row(arow, kb + 32, hi);
    }
    acc[0] = wmma_bf16(a, b0, acc[0]);
    acc[1] = wmma_bf16(a, b1, acc[1]);
    acc[2] = wmma_bf16(a, b2, acc[2]);
    acc[3] = wmma_bf16(a, b3, acc[3]);
    a = an;
  }
  // C/D layout: VGPR v -> row (v + 8*hi), col = lane&15
#pragma unroll
  for (int t = 0; t < 4; ++t) {
    int col = colbase + t * 16 + r;
    float bv = bias[wcbase + t * 16 + r];
#pragma unroll
    for (int v = 0; v < 8; ++v) {
      int grow = mtile * 16 + v + 8 * hi;
      if (grow < N) {
        float val = gelu_exact(acc[t][v] + bv);
        node[(size_t)ci[grow] * 256 + col] = val;
      }
    }
  }
}

// ---- h = x(bf16 [N,256]) @ W (via W^T bf16 [256][256]) -> f32 [N,256] ----
__global__ __launch_bounds__(256) void gemm256(
    const __bf16* __restrict__ xb, const __bf16* __restrict__ WT,
    float* __restrict__ out, int N) {
  const int wave = threadIdx.x >> 5;
  const int lane = threadIdx.x & 31;
  const int r = lane & 15, hi = lane >> 4;
  const int mtile = blockIdx.x * 2 + (wave >> 2);
  const int cb = wave & 3;
  const int row = mtile * 16 + r;
  const int rowc = (row < N) ? row : 0;
  const __bf16* arow = xb + (size_t)rowc * 256;
  const __bf16* wcol0 = WT + (size_t)(cb * 64 + r) * 256;

  v8f acc[4] = {};
  v16bf a = load_a_bf16row(arow, 0, hi);
#pragma unroll
  for (int kb = 0; kb < 256; kb += 32) {
    v16bf b0 = load_b_wt(wcol0 + (size_t)0 * 16 * 256, kb, hi);
    v16bf b1 = load_b_wt(wcol0 + (size_t)1 * 16 * 256, kb, hi);
    v16bf b2 = load_b_wt(wcol0 + (size_t)2 * 16 * 256, kb, hi);
    v16bf b3 = load_b_wt(wcol0 + (size_t)3 * 16 * 256, kb, hi);
    v16bf an = a;
    if (kb + 32 < 256) an = load_a_bf16row(arow, kb + 32, hi);
    acc[0] = wmma_bf16(a, b0, acc[0]);
    acc[1] = wmma_bf16(a, b1, acc[1]);
    acc[2] = wmma_bf16(a, b2, acc[2]);
    acc[3] = wmma_bf16(a, b3, acc[3]);
    a = an;
  }
#pragma unroll
  for (int t = 0; t < 4; ++t) {
    int col = cb * 64 + t * 16 + r;
#pragma unroll
    for (int v = 0; v < 8; ++v) {
      int grow = mtile * 16 + v + 8 * hi;
      if (grow < N) out[(size_t)grow * 256 + col] = acc[t][v];
    }
  }
}

// ---- per-node attention scalars: s = h[n] . a  (one wave32 per node) ----
__global__ __launch_bounds__(256) void node_scalars(
    const float* __restrict__ h, const float* __restrict__ asrc,
    const float* __restrict__ adst, float* __restrict__ ssrc,
    float* __restrict__ sdst, int N) {
  int wave = threadIdx.x >> 5, lane = threadIdx.x & 31;
  int node = blockIdx.x * 8 + wave;
  if (node >= N) return;
  const float* row = h + (size_t)node * 256;
  float p0 = 0.f, p1 = 0.f;
#pragma unroll
  for (int k = 0; k < 8; ++k) {
    int c = lane + 32 * k;
    float v = row[c];
    p0 += v * asrc[c];
    p1 += v * adst[c];
  }
#pragma unroll
  for (int m = 16; m > 0; m >>= 1) {
    p0 += __shfl_xor(p0, m, 32);
    p1 += __shfl_xor(p1, m, 32);
  }
  if (lane == 0) { ssrc[node] = p0; sdst[node] = p1; }
}

// ---- edge pass 1: e = leaky_relu(s_src[s]+s_dst[d]); segment max ----
__global__ void edge_max_k(const int* __restrict__ edges, int E, int N,
                           const float* __restrict__ ssrc, const float* __restrict__ sdst,
                           float* __restrict__ ebuf, float* __restrict__ mmax) {
  int i = blockIdx.x * blockDim.x + threadIdx.x;
  if (i >= E + N) return;
  int s, d;
  if (i < E) { s = edges[2 * i]; d = edges[2 * i + 1]; } else { s = d = i - E; }
  float e = ssrc[s] + sdst[d];
  e = (e > 0.f) ? e : 0.2f * e;
  ebuf[i] = e;
  atomic_max_f32(mmax + d, e);
}

// ---- edge pass 2: ex = exp(e - m[d]); segment sum ----
__global__ void edge_exp_k(const int* __restrict__ edges, int E, int N,
                           float* __restrict__ ebuf, const float* __restrict__ mmax,
                           float* __restrict__ den) {
  int i = blockIdx.x * blockDim.x + threadIdx.x;
  if (i >= E + N) return;
  int d = (i < E) ? edges[2 * i + 1] : (i - E);
  float ex = expf(ebuf[i] - mmax[d]);
  ebuf[i] = ex;
  atomicAdd(den + d, ex);
}

// ---- edge pass 3: acc[d] += (ex/den[d]) * h[s]  (one wave32 per edge) ----
__global__ __launch_bounds__(256) void edge_aggr(
    const int* __restrict__ edges, int E, int N,
    const float* __restrict__ ebuf, const float* __restrict__ den,
    const float* __restrict__ h, float* __restrict__ acc) {
  int wave = threadIdx.x >> 5, lane = threadIdx.x & 31;
  int i = blockIdx.x * 8 + wave;
  if (i >= E + N) return;
  int s, d;
  if (i < E) { s = edges[2 * i]; d = edges[2 * i + 1]; } else { s = d = i - E; }
  float alpha = ebuf[i] / den[d];
  const float4* hs4 = (const float4*)(h + (size_t)s * 256);
  float* ad = acc + (size_t)d * 256;
#pragma unroll
  for (int k = 0; k < 2; ++k) {
    int c4 = lane + 32 * k;           // 64 float4 per 256-wide row
    float4 v = hs4[c4];
    int c = c4 * 4;
    atomicAdd(ad + c + 0, alpha * v.x);
    atomicAdd(ad + c + 1, alpha * v.y);
    atomicAdd(ad + c + 2, alpha * v.z);
    atomicAdd(ad + c + 3, alpha * v.w);
  }
}

// ---- out = act(acc + bias), act: 0 = relu, 1 = exact gelu; 4 elems/thread ----
__global__ void finalize_k(const float* __restrict__ acc, const float* __restrict__ bias,
                           float* __restrict__ out, size_t n4, int act) {
  size_t i = (size_t)blockIdx.x * blockDim.x + threadIdx.x;
  if (i >= n4) return;
  float4 a = ((const float4*)acc)[i];
  int cb = ((int)(i & 63)) * 4;                 // column of first lane-element
  float4 b = *(const float4*)(bias + cb);
  float4 o;
  if (act == 0) {
    o.x = fmaxf(a.x + b.x, 0.f); o.y = fmaxf(a.y + b.y, 0.f);
    o.z = fmaxf(a.z + b.z, 0.f); o.w = fmaxf(a.w + b.w, 0.f);
  } else {
    o.x = gelu_exact(a.x + b.x); o.y = gelu_exact(a.y + b.y);
    o.z = gelu_exact(a.z + b.z); o.w = gelu_exact(a.w + b.w);
  }
  ((float4*)out)[i] = o;
}

extern "C" void kernel_launch(void* const* d_in, const int* in_sizes, int n_in,
                              void* d_out, int out_size, void* d_ws, size_t ws_size,
                              hipStream_t stream) {
  const float* img   = (const float*)d_in[0];
  const float* txt   = (const float*)d_in[1];
  const int*   ci    = (const int*)d_in[2];
  const int*   edges = (const int*)d_in[3];
  const float* w_img = (const float*)d_in[4];
  const float* b_img = (const float*)d_in[5];
  const float* w_txt = (const float*)d_in[6];
  const float* b_txt = (const float*)d_in[7];
  const float* W0    = (const float*)d_in[8];
  const float* as0   = (const float*)d_in[9];
  const float* ad0   = (const float*)d_in[10];
  const float* bi0   = (const float*)d_in[11];
  const float* W1    = (const float*)d_in[12];
  const float* as1   = (const float*)d_in[13];
  const float* ad1   = (const float*)d_in[14];
  const float* bi1   = (const float*)d_in[15];
  (void)n_in; (void)out_size; (void)ws_size;

  const int N  = in_sizes[2];
  const int E  = in_sizes[3] / 2;
  const int ET = E + N;
  const size_t nf  = (size_t)N * 256;
  const size_t nf4 = nf / 4;

  char* wsb = (char*)d_ws;
  size_t off = 0;
  auto take = [&](size_t bytes) -> void* {
    void* p = wsb + off;
    off = (off + bytes + 255) & ~(size_t)255;
    return p;
  };
  float*  bufA = (float*)take(nf * 4);
  float*  bufB = (float*)take(nf * 4);
  __bf16* xb   = (__bf16*)take(nf * 2);
  float*  ssrc = (float*)take((size_t)N * 4);
  float*  sdst = (float*)take((size_t)N * 4);
  float*  mmax = (float*)take((size_t)N * 4);
  float*  den  = (float*)take((size_t)N * 4);
  float*  ebuf = (float*)take((size_t)ET * 4);
  __bf16* wTi  = (__bf16*)take((size_t)128 * 768 * 2);
  __bf16* wTt  = (__bf16*)take((size_t)128 * 768 * 2);
  __bf16* W0T  = (__bf16*)take((size_t)256 * 256 * 2);
  __bf16* W1T  = (__bf16*)take((size_t)256 * 256 * 2);

  // weights -> bf16 transposed [col][K]
  prep_wT<<<(768 * 128 + 255) / 256, 256, 0, stream>>>(w_img, wTi, 768, 128);
  prep_wT<<<(768 * 128 + 255) / 256, 256, 0, stream>>>(w_txt, wTt, 768, 128);
  prep_wT<<<(256 * 256 + 255) / 256, 256, 0, stream>>>(W0, W0T, 256, 256);
  prep_wT<<<(256 * 256 + 255) / 256, 256, 0, stream>>>(W1, W1T, 256, 256);

  const int fb  = (int)((nf + 255) / 256);     // elementwise grid over N*256
  const int fb4 = (int)((nf4 + 255) / 256);    // vectorized x4 grid
  const int gb  = (N + 31) / 32;               // GEMM grid: 2 M-tiles per block
  const int eb  = (ET + 255) / 256;            // per-edge scalar grid
  const int wb  = (ET + 7) / 8;                // per-edge wave grid

  // node features: zero, fused upscale GEMM + GELU + scatter
  fill_f32<<<fb, 256, 0, stream>>>(bufA, 0.0f, nf);
  upscale_gemm<<<gb, 256, 0, stream>>>(img, txt, wTi, wTt, b_img, b_txt, ci, bufA, N);

  auto run_layer = [&](const float* xin, float* hbuf, float* accbuf,
                       const __bf16* WT, const float* asv, const float* adv,
                       const float* bias, float* outbuf, int act) {
    cvt_bf16_v4<<<fb4, 256, 0, stream>>>(xin, xb, nf4);
    gemm256<<<gb, 256, 0, stream>>>(xb, WT, hbuf, N);
    node_scalars<<<(N + 7) / 8, 256, 0, stream>>>(hbuf, asv, adv, ssrc, sdst, N);
    fill_f32<<<fb, 256, 0, stream>>>(accbuf, 0.0f, nf);
    fill_f32<<<(N + 255) / 256, 256, 0, stream>>>(mmax, -INFINITY, (size_t)N);
    fill_f32<<<(N + 255) / 256, 256, 0, stream>>>(den, 0.0f, (size_t)N);
    edge_max_k<<<eb, 256, 0, stream>>>(edges, E, N, ssrc, sdst, ebuf, mmax);
    edge_exp_k<<<eb, 256, 0, stream>>>(edges, E, N, ebuf, mmax, den);
    edge_aggr<<<wb, 256, 0, stream>>>(edges, E, N, ebuf, den, hbuf, accbuf);
    finalize_k<<<fb4, 256, 0, stream>>>(accbuf, bias, outbuf, nf4, act);
  };

  // layer 0: x=bufA -> h=bufB, acc reuses bufA, relu output -> bufB
  run_layer(bufA, bufB, bufA, W0T, as0, ad0, bi0, bufB, /*relu*/0);
  // layer 1: x=bufB -> h=bufA, acc reuses bufB, gelu output -> d_out
  run_layer(bufB, bufA, bufB, W1T, as1, ad1, bi1, (float*)d_out, /*gelu*/1);
}
